// Memory_4698694221840
// MI455X (gfx1250) — matrix operand order repaired and checked by reading, not compiled, not run on hardware
//
#include <hip/hip_runtime.h>
#include <hip/hip_bf16.h>
#include <math.h>
#include <stdint.h>

// ---------------------------------------------------------------------------
// NTM memory-addressing step for MI455X (gfx1250, wave32, WMMA + TDM).
// B == N == 512, M == 256, A == 128, HS == 512, C == 512.
// ---------------------------------------------------------------------------

#define NN   512
#define MM   256
#define AA   128
#define HS   512
#define CC   512
#define KIN  640
#define G3   1536

typedef __attribute__((ext_vector_type(16))) _Float16 v16h;
typedef __attribute__((ext_vector_type(8)))  float    v8f;
typedef unsigned int uintx4 __attribute__((ext_vector_type(4)));
typedef int          intx4  __attribute__((ext_vector_type(4)));
typedef int          intx8  __attribute__((ext_vector_type(8)));

#define TM 32
#define TN 64
#define KC 32

#if defined(__HIP_DEVICE_COMPILE__) && \
    __has_builtin(__builtin_amdgcn_tensor_load_to_lds) && \
    __has_builtin(__builtin_amdgcn_s_wait_tensorcnt)
#define HAVE_TDM 1
#else
#define HAVE_TDM 0
#endif

#if HAVE_TDM
// ---------------------------------------------------------------------------
// Issue one 2D TDM load: tile_h x tile_w f32 elements from a row-major tensor
// (row stride = stride_elems) into a compact LDS tile at lds_byte_off.
// D# layout per CDNA5 ISA 8.3/8.4 (group0 128b, group1 256b; groups 2/3 zero).
// ---------------------------------------------------------------------------
__device__ __forceinline__ void tdm_load_tile_f32(uint32_t lds_byte_off,
                                                  const float* gsrc,
                                                  uint32_t tile_w, uint32_t tile_h,
                                                  uint32_t tensor_w, uint32_t tensor_h,
                                                  uint32_t stride_elems) {
  uint64_t ga = (uint64_t)(uintptr_t)gsrc;
  uintx4 g0;
  g0.x = 1u;                                              // count=1, user D#
  g0.y = lds_byte_off;                                    // lds_addr
  g0.z = (uint32_t)ga;                                    // global_addr[31:0]
  g0.w = ((uint32_t)(ga >> 32) & 0x01FFFFFFu) | (2u << 30); // ga[56:32] | type=2
  intx8 g1;
  g1[0] = (int)(2u << 16);                                // data_size = 4 bytes
  g1[1] = (int)((tensor_w & 0xFFFFu) << 16);              // tensor_dim0[15:0] @63:48
  g1[2] = (int)(((tensor_w >> 16) & 0xFFFFu) | ((tensor_h & 0xFFFFu) << 16));
  g1[3] = (int)(((tensor_h >> 16) & 0xFFFFu) | ((tile_w & 0xFFFFu) << 16));
  g1[4] = (int)(tile_h & 0xFFFFu);                        // tile_dim1 (tile_dim2=0)
  g1[5] = (int)stride_elems;                              // tensor_dim0_stride lo32
  g1[6] = 0;
  g1[7] = 0;
  intx4 z4 = {};
#if __clang_major__ >= 23
  intx8 z8 = {};
  __builtin_amdgcn_tensor_load_to_lds(g0, g1, z4, z4, z8, 0);
#else
  __builtin_amdgcn_tensor_load_to_lds(g0, g1, z4, z4, 0);
#endif
}
#endif  // HAVE_TDM

// ---------------------------------------------------------------------------
// GEMM (TN): Out[m,n] = sum_k A[m,k] * W[n,k] (+ bias[n]).
// Tiles staged as f32 (TDM when available), converted to f16 at fragment
// gather, accumulated with v_wmma_f32_16x16x32_f16.
// ---------------------------------------------------------------------------
__global__ __launch_bounds__(256)
void gemm_tn_kernel(const float* __restrict__ A, const float* __restrict__ W,
                    const float* __restrict__ bias, float* __restrict__ Out,
                    int M, int N, int K) {
  __shared__ __align__(16) float Af[TM][KC];   // 4 KB
  __shared__ __align__(16) float Wf[TN][KC];   // 8 KB
  const int tid  = threadIdx.x;
  const int lane = tid & 31;
  const int wave = tid >> 5;
  const int wm   = wave >> 2;
  const int wn   = wave & 3;
  const int bm   = blockIdx.y * TM;
  const int bn   = blockIdx.x * TN;

  v8f acc = {};
  for (int k0 = 0; k0 < K; k0 += KC) {
#if HAVE_TDM
    if (wave == 0) {
      tdm_load_tile_f32((uint32_t)(uintptr_t)&Af[0][0], &A[(size_t)bm * K + k0],
                        KC, TM, (uint32_t)K, (uint32_t)M, (uint32_t)K);
      tdm_load_tile_f32((uint32_t)(uintptr_t)&Wf[0][0], &W[(size_t)bn * K + k0],
                        KC, TN, (uint32_t)K, (uint32_t)N, (uint32_t)K);
      __builtin_amdgcn_s_wait_tensorcnt(0);
    }
#else
    {  // A tile: 1024 f32 = 256 float4, one per thread.
      int r = tid >> 3, c = (tid & 7) * 4;
      *(float4*)&Af[r][c] = *(const float4*)&A[(size_t)(bm + r) * K + k0 + c];
    }
#pragma unroll
    for (int it = 0; it < 2; ++it) {  // W tile: 2048 f32 = 512 float4.
      int q = tid + it * 256;
      int r = q >> 3, c = (q & 7) * 4;
      *(float4*)&Wf[r][c] = *(const float4*)&W[(size_t)(bn + r) * K + k0 + c];
    }
#endif
    if (k0 + KC < K)  // temporal prefetch of next K-slab (global_prefetch_b8)
      __builtin_prefetch(&A[(size_t)(bm + lane) * K + k0 + KC], 0, 3);
    __syncthreads();

    // A fragment: row = wm*16 + (lane&15); Kbase = lane<16 ? 0 : 8;
    // element h -> K = Kbase + (h<8 ? h : h+8)  (two contiguous 32B runs).
    v16h a, b;
    const int arow = wm * 16 + (lane & 15);
    const int akb  = (lane < 16) ? 0 : 8;
#pragma unroll
    for (int h = 0; h < 16; ++h)
      a[h] = (_Float16)Af[arow][akb + (h < 8 ? h : h + 8)];
    // B fragment (B = W^T): N = lane&15 (+wn*16), K = (lane<16?0:8) + (h<8 ? h : h+8).
    const int bcol = wn * 16 + (lane & 15);
    const int bkb  = (lane < 16) ? 0 : 8;
#pragma unroll
    for (int h = 0; h < 16; ++h)
      b[h] = (_Float16)Wf[bcol][bkb + (h < 8 ? h : h + 8)];

    acc = __builtin_amdgcn_wmma_f32_16x16x32_f16(false, a, false, b,
                                                 (short)0, acc, false, false);
    __syncthreads();
  }

  const int row0 = bm + wm * 16 + (lane >> 4) * 8;
  const int col  = bn + wn * 16 + (lane & 15);
  const float bv = bias ? bias[col] : 0.0f;
#pragma unroll
  for (int r = 0; r < 8; ++r)
    Out[(size_t)(row0 + r) * N + col] = acc[r] + bv;
}

// ---------------------------------------------------------------------------
// GEMM (NN): Out[m,n] = sum_k A[m,k] * B[k,n].  B: [K x N] row-major f32.
// ---------------------------------------------------------------------------
__global__ __launch_bounds__(256)
void gemm_nn_kernel(const float* __restrict__ A, const float* __restrict__ B,
                    float* __restrict__ Out, int M, int N, int K) {
  __shared__ __align__(16) float Af[TM][KC];   // 4 KB
  __shared__ __align__(16) float Bf[KC][TN];   // 8 KB
  const int tid  = threadIdx.x;
  const int lane = tid & 31;
  const int wave = tid >> 5;
  const int wm   = wave >> 2;
  const int wn   = wave & 3;
  const int bm   = blockIdx.y * TM;
  const int bn   = blockIdx.x * TN;

  v8f acc = {};
  for (int k0 = 0; k0 < K; k0 += KC) {
#if HAVE_TDM
    if (wave == 0) {
      tdm_load_tile_f32((uint32_t)(uintptr_t)&Af[0][0], &A[(size_t)bm * K + k0],
                        KC, TM, (uint32_t)K, (uint32_t)M, (uint32_t)K);
      tdm_load_tile_f32((uint32_t)(uintptr_t)&Bf[0][0], &B[(size_t)k0 * N + bn],
                        TN, KC, (uint32_t)N, (uint32_t)K, (uint32_t)N);
      __builtin_amdgcn_s_wait_tensorcnt(0);
    }
#else
    {
      int r = tid >> 3, c = (tid & 7) * 4;
      *(float4*)&Af[r][c] = *(const float4*)&A[(size_t)(bm + r) * K + k0 + c];
    }
#pragma unroll
    for (int it = 0; it < 2; ++it) {  // B tile: 32x64 f32 = 512 float4.
      int q = tid + it * 256;
      int r = q >> 4, c = (q & 15) * 4;
      *(float4*)&Bf[r][c] = *(const float4*)&B[(size_t)(k0 + r) * N + bn + c];
    }
#endif
    if (k0 + KC < K)
      __builtin_prefetch(&B[(size_t)(k0 + KC + lane) * N + bn], 0, 3);
    __syncthreads();

    v16h a, b;
    const int arow = wm * 16 + (lane & 15);
    const int akb  = (lane < 16) ? 0 : 8;
#pragma unroll
    for (int h = 0; h < 16; ++h)
      a[h] = (_Float16)Af[arow][akb + (h < 8 ? h : h + 8)];
    const int bcol = wn * 16 + (lane & 15);
    const int bkb  = (lane < 16) ? 0 : 8;
#pragma unroll
    for (int h = 0; h < 16; ++h)
      b[h] = (_Float16)Bf[bkb + (h < 8 ? h : h + 8)][bcol];

    acc = __builtin_amdgcn_wmma_f32_16x16x32_f16(false, a, false, b,
                                                 (short)0, acc, false, false);
    __syncthreads();
  }

  const int row0 = bm + wm * 16 + (lane >> 4) * 8;
  const int col  = bn + wn * 16 + (lane & 15);
#pragma unroll
  for (int r = 0; r < 8; ++r)
    Out[(size_t)(row0 + r) * N + col] = acc[r];
}

// ---------------------------------------------------------------------------
// x = concat([action, hidden], axis=1) : [512, 640]
// ---------------------------------------------------------------------------
__global__ __launch_bounds__(256)
void pack_x_kernel(const float* __restrict__ action,
                   const float* __restrict__ hidden, float* __restrict__ x) {
  int idx = blockIdx.x * 256 + threadIdx.x;
  if (idx >= NN * KIN) return;
  int b = idx / KIN, c = idx - b * KIN;
  x[idx] = (c < AA) ? action[b * AA + c] : hidden[b * HS + (c - AA)];
}

// ---------------------------------------------------------------------------
// GRU single step, h0 = 0: h = (1-z) * tanh(gn + r*b_hh_n).
// ---------------------------------------------------------------------------
__global__ __launch_bounds__(256)
void gru_gates_kernel(const float* __restrict__ gi,
                      const float* __restrict__ b_hh, float* __restrict__ h) {
  int idx = blockIdx.x * 256 + threadIdx.x;
  if (idx >= NN * CC) return;
  int b = idx >> 9, c = idx & (CC - 1);
  const float* row = gi + (size_t)b * G3;
  float r = 1.0f / (1.0f + __expf(-(row[c] + b_hh[c])));
  float z = 1.0f / (1.0f + __expf(-(row[CC + c] + b_hh[CC + c])));
  float n = tanhf(row[2 * CC + c] + r * b_hh[2 * CC + c]);
  h[idx] = (1.0f - z) * n;
}

// ---------------------------------------------------------------------------
// Row L2-normalize: Y[r,:] = X[r,:] / (||X[r,:]|| + 1e-16).
// ---------------------------------------------------------------------------
__global__ __launch_bounds__(256)
void rownorm_kernel(const float* __restrict__ X, float* __restrict__ Y, int cols) {
  __shared__ float red[256];
  int r = blockIdx.x, tid = threadIdx.x;
  const float* src = X + (size_t)r * cols;
  float ss = 0.0f;
  for (int c = tid; c < cols; c += 256) { float v = src[c]; ss += v * v; }
  red[tid] = ss; __syncthreads();
  for (int s = 128; s > 0; s >>= 1) {
    if (tid < s) red[tid] += red[tid + s];
    __syncthreads();
  }
  float inv = 1.0f / (sqrtf(red[0]) + 1e-16f);
  float* dst = Y + (size_t)r * cols;
  for (int c = tid; c < cols; c += 256) dst[c] = src[c] * inv;
}

// ---------------------------------------------------------------------------
// Row softmax (axis=1), in place.
// ---------------------------------------------------------------------------
__global__ __launch_bounds__(256)
void softmax_rows_kernel(float* __restrict__ X, int cols) {
  __shared__ float red[256];
  int r = blockIdx.x, tid = threadIdx.x;
  float* row = X + (size_t)r * cols;
  float m = -3.0e38f;
  for (int c = tid; c < cols; c += 256) m = fmaxf(m, row[c]);
  red[tid] = m; __syncthreads();
  for (int s = 128; s > 0; s >>= 1) {
    if (tid < s) red[tid] = fmaxf(red[tid], red[tid + s]);
    __syncthreads();
  }
  m = red[0]; __syncthreads();
  float sum = 0.0f;
  for (int c = tid; c < cols; c += 256) {
    float e = __expf(row[c] - m);
    row[c] = e; sum += e;
  }
  red[tid] = sum; __syncthreads();
  for (int s = 128; s > 0; s >>= 1) {
    if (tid < s) red[tid] += red[tid + s];
    __syncthreads();
  }
  float inv = 1.0f / red[0];
  for (int c = tid; c < cols; c += 256) row[c] *= inv;
}

// ---------------------------------------------------------------------------
// g[b] = sigmoid(h·W_gate + b_gate); gamma[b] = softplus(h·W_gamma + b_gamma)
// ---------------------------------------------------------------------------
__global__ __launch_bounds__(256)
void gate_gamma_kernel(const float* __restrict__ h,
                       const float* __restrict__ Wg, const float* __restrict__ bg,
                       const float* __restrict__ Wm, const float* __restrict__ bm,
                       float* __restrict__ g, float* __restrict__ gamma) {
  __shared__ float r1[256];
  __shared__ float r2[256];
  int b = blockIdx.x, tid = threadIdx.x;
  const float* row = h + (size_t)b * CC;
  float d1 = 0.0f, d2 = 0.0f;
  for (int c = tid; c < CC; c += 256) {
    float hv = row[c];
    d1 += hv * Wg[c];
    d2 += hv * Wm[c];
  }
  r1[tid] = d1; r2[tid] = d2; __syncthreads();
  for (int s = 128; s > 0; s >>= 1) {
    if (tid < s) { r1[tid] += r1[tid + s]; r2[tid] += r2[tid + s]; }
    __syncthreads();
  }
  if (tid == 0) {
    g[b]     = 1.0f / (1.0f + __expf(-(r1[0] + bg[0])));
    gamma[b] = log1pf(__expf(r2[0] + bm[0]));
  }
}

// ---------------------------------------------------------------------------
// Interpolation (reference broadcast: gate indexed by ROW i), in place.
// ---------------------------------------------------------------------------
__global__ __launch_bounds__(256)
void interp_kernel(float* __restrict__ w, const float* __restrict__ prev,
                   const float* __restrict__ g) {
  int idx = blockIdx.x * 256 + threadIdx.x;
  if (idx >= NN * NN) return;
  int i = idx >> 9;
  float gv = g[i];
  w[idx] = gv * w[idx] + (1.0f - gv) * prev[idx];
}

// ---------------------------------------------------------------------------
// Circular conv per einsum('ijb,jb->ib'): out[i,b] = sum_j w[(i-j)&511,b]*s[j,b]
// ---------------------------------------------------------------------------
__global__ __launch_bounds__(256)
void circ_conv_kernel(const float* __restrict__ w, const float* __restrict__ s,
                      float* __restrict__ out) {
  __shared__ float wl[NN];
  __shared__ float sl[NN];
  int b = blockIdx.x, tid = threadIdx.x;
  for (int j = tid; j < NN; j += 256) {
    wl[j] = w[(size_t)j * NN + b];
    sl[j] = s[(size_t)j * NN + b];
  }
  __syncthreads();
  for (int i = tid; i < NN; i += 256) {
    float acc = 0.0f;
    for (int j = 0; j < NN; ++j) acc += wl[(i - j) & (NN - 1)] * sl[j];
    out[(size_t)i * NN + b] = acc;
  }
}

// ---------------------------------------------------------------------------
// Sharpen (gamma indexed by ROW i), normalize over axis 0 (per column b).
// ---------------------------------------------------------------------------
__global__ __launch_bounds__(256)
void sharpen_kernel(const float* __restrict__ w, const float* __restrict__ gamma,
                    float* __restrict__ out) {
  __shared__ float wp[NN];
  __shared__ float red[256];
  int b = blockIdx.x, tid = threadIdx.x;
  float sum = 0.0f;
  for (int i = tid; i < NN; i += 256) {
    float v = __powf(fmaxf(w[(size_t)i * NN + b], 0.0f), gamma[i]);
    wp[i] = v; sum += v;
  }
  red[tid] = sum; __syncthreads();
  for (int s = 128; s > 0; s >>= 1) {
    if (tid < s) red[tid] += red[tid + s];
    __syncthreads();
  }
  float inv = 1.0f / red[0];
  for (int i = tid; i < NN; i += 256) out[(size_t)i * NN + b] = wp[i] * inv;
}

// ---------------------------------------------------------------------------
// Host-side orchestration.
// ---------------------------------------------------------------------------
extern "C" void kernel_launch(void* const* d_in, const int* in_sizes, int n_in,
                              void* d_out, int out_size, void* d_ws, size_t ws_size,
                              hipStream_t stream) {
  const float* action  = (const float*)d_in[0];
  const float* hidden  = (const float*)d_in[1];
  const float* prev_rw = (const float*)d_in[2];
  const float* prev_ww = (const float*)d_in[3];
  const float* memory  = (const float*)d_in[4];
  const float* W_ih    = (const float*)d_in[5];
  // d_in[6] = W_hh unused (h0 == 0).
  const float* b_ih    = (const float*)d_in[7];
  const float* b_hh    = (const float*)d_in[8];
  const float* W_read  = (const float*)d_in[9];
  const float* b_read  = (const float*)d_in[10];
  const float* W_write = (const float*)d_in[11];
  const float* b_write = (const float*)d_in[12];
  const float* W_shift = (const float*)d_in[13];
  const float* b_shift = (const float*)d_in[14];
  const float* W_gamma = (const float*)d_in[15];
  const float* b_gamma = (const float*)d_in[16];
  const float* W_gate  = (const float*)d_in[17];
  const float* b_gate  = (const float*)d_in[18];

  float* ws = (float*)d_ws;
  float* x     = ws;               // 512*640
  float* gi    = ws + 327680;      // 512*1536
  float* h     = ws + 1114112;     // 512*512
  float* rk    = ws + 1376256;     // 512*256
  float* wk    = ws + 1507328;     // 512*256
  float* mn    = ws + 1638400;     // 512*256
  float* simr  = ws + 1769472;     // 512*512
  float* simw  = ws + 2031616;     // 512*512
  float* shift = ws + 2293760;     // 512*512
  float* gbuf  = ws + 2555904;     // 512
  float* gam   = ws + 2556416;     // 512
  float* cr    = ws + 2556928;     // 512*512
  float* cw    = ws + 2819072;     // 512*512 (end 3081216 floats = 12.3 MB)

  float* out      = (float*)d_out;
  float* read_vec = out;                       // [512, 256]
  float* rw_out   = out + NN * MM;             // [512, 512]
  float* ww_out   = out + NN * MM + NN * NN;   // [512, 512]

  pack_x_kernel<<<(NN * KIN + 255) / 256, 256, 0, stream>>>(action, hidden, x);

  gemm_tn_kernel<<<dim3(G3 / TN, NN / TM), 256, 0, stream>>>(x, W_ih, b_ih, gi, NN, G3, KIN);
  gru_gates_kernel<<<(NN * CC + 255) / 256, 256, 0, stream>>>(gi, b_hh, h);

  gemm_tn_kernel<<<dim3(MM / TN, NN / TM), 256, 0, stream>>>(h, W_read,  b_read,  rk,    NN, MM, CC);
  gemm_tn_kernel<<<dim3(MM / TN, NN / TM), 256, 0, stream>>>(h, W_write, b_write, wk,    NN, MM, CC);
  gemm_tn_kernel<<<dim3(NN / TN, NN / TM), 256, 0, stream>>>(h, W_shift, b_shift, shift, NN, NN, CC);
  softmax_rows_kernel<<<NN, 256, 0, stream>>>(shift, NN);

  rownorm_kernel<<<NN, 256, 0, stream>>>(memory, mn, MM);
  rownorm_kernel<<<NN, 256, 0, stream>>>(rk, rk, MM);
  rownorm_kernel<<<NN, 256, 0, stream>>>(wk, wk, MM);
  gemm_tn_kernel<<<dim3(NN / TN, NN / TM), 256, 0, stream>>>(mn, rk, nullptr, simr, NN, NN, MM);
  gemm_tn_kernel<<<dim3(NN / TN, NN / TM), 256, 0, stream>>>(mn, wk, nullptr, simw, NN, NN, MM);
  softmax_rows_kernel<<<NN, 256, 0, stream>>>(simr, NN);
  softmax_rows_kernel<<<NN, 256, 0, stream>>>(simw, NN);

  gate_gamma_kernel<<<NN, 256, 0, stream>>>(h, W_gate, b_gate, W_gamma, b_gamma, gbuf, gam);

  interp_kernel<<<(NN * NN + 255) / 256, 256, 0, stream>>>(simr, prev_rw, gbuf);
  interp_kernel<<<(NN * NN + 255) / 256, 256, 0, stream>>>(simw, prev_ww, gbuf);

  circ_conv_kernel<<<NN, 256, 0, stream>>>(simr, shift, cr);
  circ_conv_kernel<<<NN, 256, 0, stream>>>(simw, shift, cw);

  sharpen_kernel<<<NN, 256, 0, stream>>>(cr, gam, rw_out);
  sharpen_kernel<<<NN, 256, 0, stream>>>(cw, gam, ww_out);

  gemm_nn_kernel<<<dim3(MM / TN, NN / TM), 256, 0, stream>>>(rw_out, memory, read_vec, NN, MM, CC);
}
